// DeepseekV3Attention_58892591562973
// MI455X (gfx1250) — compile-verified, matrix operand order checked by reading, not run on hardware
//
#include <hip/hip_runtime.h>
#include <hip/hip_bf16.h>

// ---------------------------------------------------------------------------
// DeepSeek-V3 MLA attention for MI455X (gfx1250, wave32, WMMA + TDM).
// All matmuls via v_wmma_f32_16x16x32_bf16 (f32 accumulate); tile staging via
// tensor_load_to_lds (Tensor Data Mover) where available.
// ---------------------------------------------------------------------------

typedef __attribute__((ext_vector_type(16))) __bf16 v16bf;
typedef __attribute__((ext_vector_type(8)))  float  v8f;
typedef __attribute__((ext_vector_type(4)))  unsigned int u32x4;
typedef __attribute__((ext_vector_type(8)))  int    i32x8;
typedef __attribute__((ext_vector_type(4)))  int    i32x4;
typedef unsigned short u16;
typedef unsigned int   u32;
typedef unsigned long long u64;

// Problem constants
static constexpr int Bc   = 2;
static constexpr int Sc   = 2048;
static constexpr int Hc   = 2048;
static constexpr int NHc  = 16;
static constexpr int QLORA = 1536;
static constexpr int KVLORA = 512;
static constexpr int DNOPE = 128;
static constexpr int DROPE = 64;
static constexpr int DVv   = 128;
static constexpr int DQK   = 192;   // DNOPE + DROPE
static constexpr int MROWS = Bc * Sc;  // 4096
static constexpr float SCALE = 0.072168783648703220563f; // 192^-0.5

union V8 { v8f v; float f[8]; };

__device__ __forceinline__ u16 f2bf(float x) {
    union { float f; u32 u; } v; v.f = x;
    u32 r = v.u + 0x7FFFu + ((v.u >> 16) & 1u);
    return (u16)(r >> 16);
}

__device__ __forceinline__ v8f wmma_bf16(v16bf a, v16bf b, v8f c) {
    // (neg_a, A, neg_b, B, c_mod, C, reuse_a, reuse_b)
    return __builtin_amdgcn_wmma_f32_16x16x32_bf16(false, a, false, b, (short)0, c,
                                                   false, false);
}

// ---------------------------------------------------------------------------
// Tensor Data Mover: 2-D tile load Global -> LDS.
// D# per CDNA5 ISA:
//   group0: [1:0]=count=1, [63:32]=lds_addr, [120:64]=global_addr, [127:126]=2
//   group1: [17:16]=data_size(2 -> 4B), tensor_dim0/1, tile_dim0/1,
//           tensor_dim0_stride (48b); dims in dword units here.
// ---------------------------------------------------------------------------
#if __has_builtin(__builtin_amdgcn_tensor_load_to_lds)
#define HAVE_TDM 1
__device__ __forceinline__ void tdm_load_2d(u32 lds_addr, const void* gptr,
                                            u32 tile_w_dw, u32 tile_h,
                                            u32 stride_dw) {
    const u64 ga = (u64)(uintptr_t)gptr;
    const u32 td0 = stride_dw;        // tensor width >= tile width
    const u32 td1 = 0x00FFFFFFu;      // large row count (OOB never trips)

    u32x4 g0;
    g0[0] = 1u;                                   // count = 1 valid descriptor
    g0[1] = lds_addr;                             // LDS byte address
    g0[2] = (u32)ga;                              // global addr [31:0]
    g0[3] = (u32)(ga >> 32) | (2u << 30);         // global addr [56:32], type=2

    i32x8 g1;
    g1[0] = (int)(2u << 16);                      // data_size = 4 bytes
    g1[1] = (int)((td0 & 0xFFFFu) << 16);         // tensor_dim0 [15:0]
    g1[2] = (int)((td0 >> 16) | ((td1 & 0xFFFFu) << 16));   // td0 hi, td1 lo
    g1[3] = (int)((td1 >> 16) | (tile_w_dw << 16));         // td1 hi, tile_dim0
    g1[4] = (int)(tile_h & 0xFFFFu);              // tile_dim1 (tile_dim2 = 0)
    g1[5] = (int)stride_dw;                       // tensor_dim0_stride [31:0]
    g1[6] = 0;                                    // stride hi, dim1_stride lo
    g1[7] = 0;

#if defined(__clang_major__) && (__clang_major__ >= 23)
    __builtin_amdgcn_tensor_load_to_lds(g0, g1, (i32x4)0, (i32x4)0, (i32x8)0, 0);
#else
    __builtin_amdgcn_tensor_load_to_lds(g0, g1, (i32x4)0, (i32x4)0, 0);
#endif
}
#define TDM_WAIT() __builtin_amdgcn_s_wait_tensorcnt(0)
#else
#define HAVE_TDM 0
#endif

// ---------------------------------------------------------------------------
// fp32 -> bf16 conversion
// ---------------------------------------------------------------------------
__global__ void convert_f32_bf16(const float* __restrict__ in,
                                 u16* __restrict__ out, size_t n) {
    size_t i = (size_t)blockIdx.x * blockDim.x + threadIdx.x;
    if (i < n) out[i] = f2bf(in[i]);
}

// ---------------------------------------------------------------------------
// Tiled bf16 WMMA GEMM: C[M,N] = A[M,K] * W[N,K]^T
// block = 256 threads (8 waves); tile = 128(M) x 64(N); K-step = 32.
// grid = (N/64, M/128). Tiles staged via TDM when available.
// ---------------------------------------------------------------------------
template <bool OUT_BF16>
__global__ __launch_bounds__(256)
void gemm_bf16_wmma(const u16* __restrict__ A, const u16* __restrict__ W,
                    void* __restrict__ Cout, int M, int N, int K) {
    __shared__ __align__(32) u16 sA[128 * 32];
    __shared__ __align__(32) u16 sB[64 * 32];

    const int tid  = threadIdx.x;
    const int lane = tid & 31;
    const int wave = tid >> 5;
    const int wm   = wave & 3;
    const int wn   = wave >> 2;
    const int lm   = lane & 15;
    const int hf   = lane >> 4;

    const int m0 = blockIdx.y * 128;
    const int n0 = blockIdx.x * 64;

    V8 c00, c01, c10, c11;
#pragma unroll
    for (int i = 0; i < 8; ++i) { c00.f[i] = 0.f; c01.f[i] = 0.f; c10.f[i] = 0.f; c11.f[i] = 0.f; }

    for (int k0 = 0; k0 < K; k0 += 32) {
#if HAVE_TDM
        // TDM DMA: A tile 128 rows x 16 dwords, W tile 64 rows x 16 dwords
        if (tid == 0) {
            tdm_load_2d((u32)(uintptr_t)sA, A + (size_t)m0 * K + k0,
                        /*tile_w_dw=*/16, /*tile_h=*/128, /*stride_dw=*/(u32)(K >> 1));
            tdm_load_2d((u32)(uintptr_t)sB, W + (size_t)n0 * K + k0,
                        16, 64, (u32)(K >> 1));
        }
        if (k0 + 32 < K) {
            __builtin_prefetch(A + (size_t)(m0 + (tid & 127)) * K + k0 + 32, 0, 0);
            if (tid < 64)
                __builtin_prefetch(W + (size_t)(n0 + tid) * K + k0 + 32, 0, 0);
        }
        if (tid < 32) TDM_WAIT();   // issuing wave drains TENSORcnt pre-barrier
#else
#pragma unroll
        for (int it = 0; it < 2; ++it) {
            int idx   = tid + it * 256;
            int row   = idx >> 2;
            int chunk = idx & 3;
            ((uint4*)sA)[idx] =
                *(const uint4*)(A + (size_t)(m0 + row) * K + k0 + chunk * 8);
        }
        {
            int row   = tid >> 2;
            int chunk = tid & 3;
            ((uint4*)sB)[tid] =
                *(const uint4*)(W + (size_t)(n0 + row) * K + k0 + chunk * 8);
        }
        if (k0 + 32 < K) {
            __builtin_prefetch(A + (size_t)(m0 + (tid & 127)) * K + k0 + 32, 0, 0);
            if (tid < 64)
                __builtin_prefetch(W + (size_t)(n0 + tid) * K + k0 + 32, 0, 0);
        }
#endif
        __syncthreads();

        v16bf a0 = *(const v16bf*)(sA + (wm * 32 + lm) * 32 + hf * 16);
        v16bf a1 = *(const v16bf*)(sA + (wm * 32 + 16 + lm) * 32 + hf * 16);
        v16bf b0 = *(const v16bf*)(sB + (wn * 32 + lm) * 32 + hf * 16);
        v16bf b1 = *(const v16bf*)(sB + (wn * 32 + 16 + lm) * 32 + hf * 16);

        c00.v = wmma_bf16(a0, b0, c00.v);
        c01.v = wmma_bf16(a0, b1, c01.v);
        c10.v = wmma_bf16(a1, b0, c10.v);
        c11.v = wmma_bf16(a1, b1, c11.v);

        __syncthreads();
    }

    // D layout: VGPR r, lanes 0-15 -> M=r, lanes 16-31 -> M=r+8; N col = lm
    const int mBase = m0 + wm * 32;
    const int nBase = n0 + wn * 32;
#pragma unroll
    for (int r = 0; r < 8; ++r) {
        int rr0 = mBase + r + 8 * hf;
        int rr1 = rr0 + 16;
        int cc0 = nBase + lm;
        int cc1 = cc0 + 16;
        if (OUT_BF16) {
            u16* C = (u16*)Cout;
            C[(size_t)rr0 * N + cc0] = f2bf(c00.f[r]);
            C[(size_t)rr0 * N + cc1] = f2bf(c01.f[r]);
            C[(size_t)rr1 * N + cc0] = f2bf(c10.f[r]);
            C[(size_t)rr1 * N + cc1] = f2bf(c11.f[r]);
        } else {
            float* C = (float*)Cout;
            C[(size_t)rr0 * N + cc0] = c00.f[r];
            C[(size_t)rr0 * N + cc1] = c01.f[r];
            C[(size_t)rr1 * N + cc0] = c10.f[r];
            C[(size_t)rr1 * N + cc1] = c11.f[r];
        }
    }
}

// ---------------------------------------------------------------------------
// RMS norm (fp32 in, bf16 out): one block (256 thr) per row.
// ---------------------------------------------------------------------------
__global__ __launch_bounds__(256)
void rmsnorm_bf16(const float* __restrict__ in, const float* __restrict__ w,
                  u16* __restrict__ out, int D, int in_stride, int out_stride) {
    __shared__ float red[8];
    const int row = blockIdx.x;
    const float* x = in + (size_t)row * in_stride;

    float ss = 0.f;
    for (int i = threadIdx.x; i < D; i += 256) { float v = x[i]; ss += v * v; }
#pragma unroll
    for (int m = 16; m >= 1; m >>= 1) ss += __shfl_xor(ss, m, 32);
    if ((threadIdx.x & 31) == 0) red[threadIdx.x >> 5] = ss;
    __syncthreads();
    if (threadIdx.x == 0) {
        float t = 0.f;
#pragma unroll
        for (int i = 0; i < 8; ++i) t += red[i];
        red[0] = t;
    }
    __syncthreads();
    const float rinv = rsqrtf(red[0] / (float)D + 1e-6f);
    for (int i = threadIdx.x; i < D; i += 256)
        out[(size_t)row * out_stride + i] = f2bf(x[i] * rinv * w[i]);
}

// ---------------------------------------------------------------------------
// Q prep: fp32 q [B*S, NH*192] -> bf16 Q [b,h,s,192] with RoPE on d>=128.
// ---------------------------------------------------------------------------
__global__ void prep_q_kernel(const float* __restrict__ q,
                              const float* __restrict__ cosv,
                              const float* __restrict__ sinv,
                              u16* __restrict__ Q) {
    size_t idx = (size_t)blockIdx.x * blockDim.x + threadIdx.x;
    const size_t total = (size_t)MROWS * NHc * DQK;
    if (idx >= total) return;
    int d      = (int)(idx % DQK);
    size_t t   = idx / DQK;
    int h      = (int)(t % NHc);
    size_t bs  = t / NHc;             // b*S + s
    int s      = (int)(bs % Sc);
    int b      = (int)(bs / Sc);

    float x = q[idx];
    float val;
    if (d < DNOPE) {
        val = x;
    } else {
        int dr = d - DNOPE;
        float c  = cosv[bs * DROPE + dr];
        float sn = sinv[bs * DROPE + dr];
        float other = (dr < 32) ? -q[idx + 32] : q[idx - 32];
        val = x * c + other * sn;
    }
    Q[((((size_t)b * NHc + h) * Sc) + s) * DQK + d] = f2bf(val);
}

// ---------------------------------------------------------------------------
// KV prep: fp32 kv [B*S, NH*256], fp32 ckv [B*S, 576] ->
//   bf16 K [b,h,s,192] (k_nope ++ roped k_pe broadcast), bf16 V [b,h,s,128].
// ---------------------------------------------------------------------------
__global__ void prep_kv_kernel(const float* __restrict__ kv,
                               const float* __restrict__ ckv,
                               const float* __restrict__ cosv,
                               const float* __restrict__ sinv,
                               u16* __restrict__ Kd, u16* __restrict__ Vd) {
    size_t idx = (size_t)blockIdx.x * blockDim.x + threadIdx.x;
    const size_t total = (size_t)MROWS * NHc * 320;
    if (idx >= total) return;
    int d      = (int)(idx % 320);
    size_t t   = idx / 320;
    int h      = (int)(t % NHc);
    size_t bs  = t / NHc;
    int s      = (int)(bs % Sc);
    int b      = (int)(bs / Sc);
    size_t bhs = (((size_t)b * NHc + h) * Sc) + s;

    if (d < DNOPE) {
        float x = kv[bs * (size_t)(NHc * 256) + (size_t)h * 256 + d];
        Kd[bhs * DQK + d] = f2bf(x);
    } else if (d < DQK) {
        int dr = d - DNOPE;
        const float* kp = ckv + bs * 576 + KVLORA;
        float x  = kp[dr];
        float c  = cosv[bs * DROPE + dr];
        float sn = sinv[bs * DROPE + dr];
        float other = (dr < 32) ? -kp[dr + 32] : kp[dr - 32];
        Kd[bhs * DQK + d] = f2bf(x * c + other * sn);
    } else {
        int dv = d - DQK;
        float x = kv[bs * (size_t)(NHc * 256) + (size_t)h * 256 + DNOPE + dv];
        Vd[bhs * DVv + dv] = f2bf(x);
    }
}

// ---------------------------------------------------------------------------
// Causal flash attention, bf16 WMMA, fp32 online softmax.
// grid = (S/64, B*NH); block = 128 (4 waves). Wave owns 16 q-rows.
// K tile staged via TDM; V staged transposed (d-major) for P.V B-fragments.
// ---------------------------------------------------------------------------
__global__ __launch_bounds__(128)
void mla_flash_attn(const u16* __restrict__ Q, const u16* __restrict__ Kd,
                    const u16* __restrict__ Vd, u16* __restrict__ O) {
    __shared__ __align__(32) u16 sK[64 * DQK];     // [k][d]   24.6 KB
    __shared__ __align__(32) u16 sVt[DVv * 64];    // [d][k]   16 KB
    __shared__ __align__(32) u16 sP[4 * 16 * 64];  // per-wave P tile, 8 KB

    const int tid  = threadIdx.x;
    const int lane = tid & 31;
    const int wave = tid >> 5;
    const int lm   = lane & 15;
    const int hf   = lane >> 4;

    const int bh  = blockIdx.y;          // b*NH + h
    const int b   = bh >> 4;
    const int h   = bh & 15;
    const int qb  = blockIdx.x;          // q block of 64
    const int qr0 = qb * 64 + wave * 16; // this wave's first q row

    // Q fragments: 6 x (16x32) bf16 A-tiles in registers
    v16bf qf[6];
#pragma unroll
    for (int kk = 0; kk < 6; ++kk)
        qf[kk] = *(const v16bf*)(Q + ((size_t)bh * Sc + qr0 + lm) * DQK +
                                 kk * 32 + hf * 16);

    V8 acc[8];
#pragma unroll
    for (int dtile = 0; dtile < 8; ++dtile)
#pragma unroll
        for (int r = 0; r < 8; ++r) acc[dtile].f[r] = 0.f;

    float mrow[8], lrow[8];
#pragma unroll
    for (int r = 0; r < 8; ++r) { mrow[r] = -3.0e38f; lrow[r] = 0.f; }

    for (int kb = 0; kb <= qb; ++kb) {
        // ---- stage K block [64 x 192] bf16 (rows contiguous, 96 dw each) ----
#if HAVE_TDM
        if (tid == 0)
            tdm_load_2d((u32)(uintptr_t)sK,
                        Kd + ((size_t)bh * Sc + kb * 64) * DQK,
                        /*tile_w_dw=*/96, /*tile_h=*/64, /*stride_dw=*/96);
#else
#pragma unroll
        for (int it = 0; it < 12; ++it) {
            int idx = tid + it * 128;       // 1536 uint4 total
            int row = idx / 24;
            int ch  = idx % 24;
            ((uint4*)sK)[idx] = *(const uint4*)(
                Kd + ((size_t)bh * Sc + kb * 64 + row) * DQK + ch * 8);
        }
#endif
        // ---- stage V block transposed: sVt[d][k] ----
#pragma unroll
        for (int it = 0; it < 8; ++it) {
            int e  = tid + it * 128;        // 1024 uint4 total
            int kr = e >> 4;
            int db = e & 15;
            uint4 u = *(const uint4*)(
                Vd + ((size_t)bh * Sc + kb * 64 + kr) * DVv + db * 8);
            const u16* p = (const u16*)&u;
#pragma unroll
            for (int j = 0; j < 8; ++j) sVt[(db * 8 + j) * 64 + kr] = p[j];
        }
#if HAVE_TDM
        if (tid < 32) TDM_WAIT();
#endif
        __syncthreads();

        // ---- scores: 4 tiles of 16x16, K-dim 192 = 6 wmma each ----
        V8 st[4];
#pragma unroll
        for (int t4 = 0; t4 < 4; ++t4) {
#pragma unroll
            for (int r = 0; r < 8; ++r) st[t4].f[r] = 0.f;
#pragma unroll
            for (int kk = 0; kk < 6; ++kk) {
                v16bf bk = *(const v16bf*)(sK + (t4 * 16 + lm) * DQK +
                                           kk * 32 + hf * 16);
                st[t4].v = wmma_bf16(qf[kk], bk, st[t4].v);
            }
        }

        // ---- scale + causal mask + row max ----
        float rmax[8];
#pragma unroll
        for (int r = 0; r < 8; ++r) rmax[r] = -3.0e38f;
#pragma unroll
        for (int t4 = 0; t4 < 4; ++t4) {
            int kc = kb * 64 + t4 * 16 + lm;
#pragma unroll
            for (int r = 0; r < 8; ++r) {
                int qr = qr0 + r + 8 * hf;
                float x = st[t4].f[r] * SCALE;
                if (kc > qr) x = -3.0e38f;
                st[t4].f[r] = x;
                rmax[r] = fmaxf(rmax[r], x);
            }
        }
#pragma unroll
        for (int r = 0; r < 8; ++r) {
#pragma unroll
            for (int m = 8; m >= 1; m >>= 1)
                rmax[r] = fmaxf(rmax[r], __shfl_xor(rmax[r], m, 32));
        }

        // ---- online softmax update ----
        float alpha[8], rsum[8];
#pragma unroll
        for (int r = 0; r < 8; ++r) {
            float mn = fmaxf(mrow[r], rmax[r]);
            alpha[r] = __expf(mrow[r] - mn);
            mrow[r]  = mn;
            rsum[r]  = 0.f;
        }
#pragma unroll
        for (int t4 = 0; t4 < 4; ++t4) {
#pragma unroll
            for (int r = 0; r < 8; ++r) {
                float p = __expf(st[t4].f[r] - mrow[r]);
                rsum[r] += p;
                sP[wave * 1024 + (r + 8 * hf) * 64 + t4 * 16 + lm] = f2bf(p);
            }
        }
#pragma unroll
        for (int r = 0; r < 8; ++r) {
#pragma unroll
            for (int m = 8; m >= 1; m >>= 1)
                rsum[r] += __shfl_xor(rsum[r], m, 32);
            lrow[r] = lrow[r] * alpha[r] + rsum[r];
        }
#pragma unroll
        for (int dtile = 0; dtile < 8; ++dtile)
#pragma unroll
            for (int r = 0; r < 8; ++r) acc[dtile].f[r] *= alpha[r];

        __syncthreads();  // P visible across lanes

        // ---- P (16x64) x V (64x128): 2 K-steps x 8 d-tiles ----
#pragma unroll
        for (int kk = 0; kk < 2; ++kk) {
            v16bf pa = *(const v16bf*)(sP + wave * 1024 + lm * 64 +
                                       kk * 32 + hf * 16);
#pragma unroll
            for (int dtile = 0; dtile < 8; ++dtile) {
                v16bf bv = *(const v16bf*)(sVt + (dtile * 16 + lm) * 64 +
                                           kk * 32 + hf * 16);
                acc[dtile].v = wmma_bf16(pa, bv, acc[dtile].v);
            }
        }
        __syncthreads();  // before restaging sK/sVt
    }

    // ---- epilogue: normalize, write bf16 O[b*S+s, h*128+d] ----
#pragma unroll
    for (int dtile = 0; dtile < 8; ++dtile) {
#pragma unroll
        for (int r = 0; r < 8; ++r) {
            int qr = qr0 + r + 8 * hf;
            float v = acc[dtile].f[r] / lrow[r];
            O[((size_t)b * Sc + qr) * (NHc * DVv) + h * DVv + dtile * 16 + lm] =
                f2bf(v);
        }
    }
}

// ---------------------------------------------------------------------------
// Host orchestration
// ---------------------------------------------------------------------------
extern "C" void kernel_launch(void* const* d_in, const int* in_sizes, int n_in,
                              void* d_out, int out_size, void* d_ws, size_t ws_size,
                              hipStream_t stream) {
    (void)in_sizes; (void)n_in; (void)out_size; (void)ws_size;

    const float* hidden  = (const float*)d_in[0];
    const float* cosv    = (const float*)d_in[1];
    const float* sinv    = (const float*)d_in[2];
    const float* q_a_w   = (const float*)d_in[3];
    const float* q_a_ln  = (const float*)d_in[4];
    const float* q_b_w   = (const float*)d_in[5];
    const float* kv_a_w  = (const float*)d_in[6];
    const float* kv_a_ln = (const float*)d_in[7];
    const float* kv_b_w  = (const float*)d_in[8];
    const float* o_w     = (const float*)d_in[9];

    char* ws = (char*)d_ws;
    size_t off = 0;
    auto carve = [&](size_t bytes) -> char* {
        char* p = ws + off;
        off = (off + bytes + 255) & ~(size_t)255;
        return p;
    };

    u16* hid_bf   = (u16*)carve((size_t)MROWS * Hc * 2);
    u16* wqa_bf   = (u16*)carve((size_t)QLORA * Hc * 2);
    u16* wqb_bf   = (u16*)carve((size_t)NHc * DQK * QLORA * 2);
    u16* wkva_bf  = (u16*)carve((size_t)(KVLORA + DROPE) * Hc * 2);
    u16* wkvb_bf  = (u16*)carve((size_t)NHc * (DNOPE + DVv) * KVLORA * 2);
    u16* wo_bf    = (u16*)carve((size_t)Hc * NHc * DVv * 2);
    float* qa_f   = (float*)carve((size_t)MROWS * QLORA * 4);
    u16* qa_n     = (u16*)carve((size_t)MROWS * QLORA * 2);
    float* ckv_f  = (float*)carve((size_t)MROWS * (KVLORA + DROPE) * 4);
    u16* ckv_n    = (u16*)carve((size_t)MROWS * KVLORA * 2);
    float* q_f    = (float*)carve((size_t)MROWS * NHc * DQK * 4);
    float* kv_f   = (float*)carve((size_t)MROWS * NHc * (DNOPE + DVv) * 4);
    u16* Qbf      = (u16*)carve((size_t)Bc * NHc * Sc * DQK * 2);
    u16* Kbf      = (u16*)carve((size_t)Bc * NHc * Sc * DQK * 2);
    u16* Vbf      = (u16*)carve((size_t)Bc * NHc * Sc * DVv * 2);
    u16* attn_bf  = (u16*)carve((size_t)MROWS * NHc * DVv * 2);

    auto cvt = [&](const float* src, u16* dst, size_t n) {
        convert_f32_bf16<<<(unsigned)((n + 255) / 256), 256, 0, stream>>>(src, dst, n);
    };
    cvt(hidden, hid_bf,  (size_t)MROWS * Hc);
    cvt(q_a_w,  wqa_bf,  (size_t)QLORA * Hc);
    cvt(q_b_w,  wqb_bf,  (size_t)NHc * DQK * QLORA);
    cvt(kv_a_w, wkva_bf, (size_t)(KVLORA + DROPE) * Hc);
    cvt(kv_b_w, wkvb_bf, (size_t)NHc * (DNOPE + DVv) * KVLORA);
    cvt(o_w,    wo_bf,   (size_t)Hc * NHc * DVv);

    // q_a = hidden @ q_a_w.T   [4096,1536]
    gemm_bf16_wmma<false><<<dim3(QLORA / 64, MROWS / 128), 256, 0, stream>>>(
        hid_bf, wqa_bf, qa_f, MROWS, QLORA, Hc);
    // ckv = hidden @ kv_a_w.T  [4096,576]
    gemm_bf16_wmma<false><<<dim3((KVLORA + DROPE) / 64, MROWS / 128), 256, 0, stream>>>(
        hid_bf, wkva_bf, ckv_f, MROWS, KVLORA + DROPE, Hc);

    rmsnorm_bf16<<<MROWS, 256, 0, stream>>>(qa_f, q_a_ln, qa_n, QLORA, QLORA, QLORA);
    rmsnorm_bf16<<<MROWS, 256, 0, stream>>>(ckv_f, kv_a_ln, ckv_n, KVLORA,
                                            KVLORA + DROPE, KVLORA);

    // q = rmsnorm(q_a) @ q_b_w.T  [4096, 3072]
    gemm_bf16_wmma<false><<<dim3(NHc * DQK / 64, MROWS / 128), 256, 0, stream>>>(
        qa_n, wqb_bf, q_f, MROWS, NHc * DQK, QLORA);
    // kv = rmsnorm(ckv) @ kv_b_w.T [4096, 4096]
    gemm_bf16_wmma<false><<<dim3(NHc * (DNOPE + DVv) / 64, MROWS / 128), 256, 0, stream>>>(
        ckv_n, wkvb_bf, kv_f, MROWS, NHc * (DNOPE + DVv), KVLORA);

    {
        size_t total = (size_t)MROWS * NHc * DQK;
        prep_q_kernel<<<(unsigned)((total + 255) / 256), 256, 0, stream>>>(
            q_f, cosv, sinv, Qbf);
    }
    {
        size_t total = (size_t)MROWS * NHc * 320;
        prep_kv_kernel<<<(unsigned)((total + 255) / 256), 256, 0, stream>>>(
            kv_f, ckv_f, cosv, sinv, Kbf, Vbf);
    }

    mla_flash_attn<<<dim3(Sc / 64, Bc * NHc), 128, 0, stream>>>(Qbf, Kbf, Vbf, attn_bf);

    // out = attn @ o_w.T  [4096, 2048] fp32 -> d_out
    gemm_bf16_wmma<false><<<dim3(Hc / 64, MROWS / 128), 256, 0, stream>>>(
        attn_bf, wo_bf, (float*)d_out, MROWS, Hc, NHc * DVv);
}
